// SE3Layer_79748952752292
// MI455X (gfx1250) — matrix operand order repaired and checked by reading, not compiled, not run on hardware
//
#include <hip/hip_runtime.h>

typedef __bf16 bf16_t;
typedef __attribute__((ext_vector_type(16))) __bf16 v16bf;
typedef __attribute__((ext_vector_type(8)))  __bf16 v8bf;
typedef __attribute__((ext_vector_type(8)))  float  v8f;

#define DD 128   // feature dim
#define KC 4     // K chunks of 32 (128 / 32)
#define MT 4     // M-tiles (of 16 rows) per block
#define RPB (MT * 16)   // rows per block = 64

// ---------------------------------------------------------------------------
// WMMA fragment helpers (wave32, 16x16x32 bf16, fp32 accumulate).
// A tile: 16x128 bf16 row-major in LDS.
//   lane L -> row = L&15, khalf = L>>4; element j holds
//   K = kc*32 + khalf*16 + j  => one contiguous 32B run.
// B weights pre-packed so each lane reads 16 contiguous bf16 per K-chunk:
//   pk[((ntile*KC+kc)*32+lane)*16+j] = W[kc*32+(lane>>4)*16+j][ntile*16+(lane&15)]
// ---------------------------------------------------------------------------
__device__ __forceinline__ v16bf a_frag(const bf16_t* __restrict__ tile, int lane, int kc) {
  const int r  = lane & 15;
  const int kh = lane >> 4;
  const bf16_t* p = tile + r * DD + kc * 32 + kh * 16;
  v8bf lo = *(const v8bf*)(p);
  v8bf hi = *(const v8bf*)(p + 8);
  return __builtin_shufflevector(lo, hi, 0,1,2,3,4,5,6,7,8,9,10,11,12,13,14,15);
}

__device__ __forceinline__ v16bf b_frag(const bf16_t* __restrict__ wpk, int lane, int ntile, int kc) {
  return *(const v16bf*)(wpk + ((((ntile * KC) + kc) * 32 + lane) << 4));
}

// Preload a wave's 4 B fragments (one 128x128 weight, fixed N-tile) to registers.
__device__ __forceinline__ void load_bregs(const bf16_t* __restrict__ wpk,
                                           int lane, int ntile, v16bf* breg) {
#pragma unroll
  for (int kc = 0; kc < KC; ++kc) breg[kc] = b_frag(wpk, lane, ntile, kc);
}

// acc += A(16x128, LDS) * W(128x128, register B fragments), this wave's N-tile
__device__ __forceinline__ v8f gemm_regB(const bf16_t* __restrict__ Atile,
                                         const v16bf* __restrict__ breg,
                                         int lane, v8f acc) {
#pragma unroll
  for (int kc = 0; kc < KC; ++kc) {
    v16bf a = a_frag(Atile, lane, kc);
    acc = __builtin_amdgcn_wmma_f32_16x16x32_bf16(false, a, false, breg[kc],
                                                  (short)0, acc, false, false);
  }
  return acc;
}

// Per-row mean / rstd over a 16x128 fp32 LDS tile. 16 threads per row
// (consecutive tids -> same 16-lane half-wave), shfl_xor reduction.
__device__ __forceinline__ void row_stats(const float* __restrict__ t,
                                          float* __restrict__ mu, float* __restrict__ rs,
                                          int tid) {
  const int row = tid >> 4, seg = tid & 15;
  float s = 0.f, ss = 0.f;
  const float* p = t + row * DD + seg * 8;
#pragma unroll
  for (int j = 0; j < 8; ++j) { float v = p[j]; s += v; ss += v * v; }
#pragma unroll
  for (int o = 8; o >= 1; o >>= 1) {
    s  += __shfl_xor(s,  o, 16);
    ss += __shfl_xor(ss, o, 16);
  }
  if (seg == 0) {
    float m   = s  * (1.0f / 128.0f);
    float var = ss * (1.0f / 128.0f) - m * m;
    mu[row] = m;
    rs[row] = rsqrtf(var + 1e-5f);
  }
}

// ---------------------------------------------------------------------------
// Weight re-pack: 7 blocks of 128x128 fp32 -> bf16 in WMMA-B layout.
//  blk0 node_w1 | blk1 node_w2 | blk2 edge_w1[0:128] | blk3 edge_w1[128:256]
//  blk4 edge_w2 | blk5 upd_w[0:128] | blk6 upd_w[128:256]
// ---------------------------------------------------------------------------
__global__ __launch_bounds__(256) void pack_weights_kernel(
    const float* __restrict__ node_w1, const float* __restrict__ node_w2,
    const float* __restrict__ edge_w1, const float* __restrict__ edge_w2,
    const float* __restrict__ upd_w, bf16_t* __restrict__ packed) {
  int idx = blockIdx.x * 256 + threadIdx.x;       // 0 .. 7*16384-1
  int blk = idx >> 14;
  int p   = idx & 16383;
  int j = p & 15, lane = (p >> 4) & 31, kc = (p >> 9) & 3, ntile = p >> 11;
  int k = kc * 32 + ((lane >> 4) << 4) + j;
  int n = (ntile << 4) + (lane & 15);
  const float* src = node_w1;
  int rowOff = 0;
  switch (blk) {
    case 0: src = node_w1; break;
    case 1: src = node_w2; break;
    case 2: src = edge_w1; break;
    case 3: src = edge_w1; rowOff = 128; break;
    case 4: src = edge_w2; break;
    case 5: src = upd_w;   break;
    default: src = upd_w;  rowOff = 128; break;
  }
  packed[idx] = (bf16_t)src[(size_t)(k + rowOff) * DD + n];
}

// ---------------------------------------------------------------------------
// Node MLP: h_nodes = relu(LN(x@W1+b1)) @ W2 + b2.  Also emits x in bf16.
// Block = 256 threads = 8 waves; 4 M-tiles (64 nodes) per block; wave w owns
// N-tile w with both weight matrices' B-fragments register-resident.
// ---------------------------------------------------------------------------
__global__ __launch_bounds__(256) void node_mlp_kernel(
    const float* __restrict__ x,
    const bf16_t* __restrict__ w1pk, const float* __restrict__ b1,
    const float* __restrict__ g1, const float* __restrict__ be1,
    const bf16_t* __restrict__ w2pk, const float* __restrict__ b2,
    float* __restrict__ h_nodes, bf16_t* __restrict__ x_bf, int N) {
  __shared__ __align__(16) bf16_t As[16 * DD];
  __shared__ __align__(16) float  Ts[16 * DD];
  __shared__ __align__(16) bf16_t A2[16 * DD];
  __shared__ float mu[16], rs[16];

  const int tid  = threadIdx.x;
  const int lane = tid & 31, wave = tid >> 5;
  const int row  = tid >> 4, seg  = tid & 15;
  const int col   = wave * 16 + (lane & 15);
  const int rbase = (lane >> 4) * 8;
  const int nBase = blockIdx.x * RPB;

  v16bf bw1[KC], bw2[KC];
  load_bregs(w1pk, lane, wave, bw1);
  load_bregs(w2pk, lane, wave, bw2);

  const float bb1 = b1[col];
  const float bb2 = b2[col];

  for (int t = 0; t < MT; ++t) {
    const int tBase = nBase + t * 16;
    const int gr = tBase + row;
    __syncthreads();   // LDS buffers from previous tile are dead
    { // load x tile, convert to bf16 into LDS + global bf16 table
      uint4 u;
      bf16_t* hp = (bf16_t*)&u;
      if (gr < N) {
        const float* xp = x + (size_t)gr * DD + seg * 8;
#pragma unroll
        for (int j = 0; j < 8; ++j) hp[j] = (bf16_t)xp[j];
      } else {
#pragma unroll
        for (int j = 0; j < 8; ++j) hp[j] = (bf16_t)0.0f;
      }
      *(uint4*)(As + row * DD + seg * 8) = u;
      if (gr < N) *(uint4*)(x_bf + (size_t)gr * DD + seg * 8) = u;
    }
    __syncthreads();

    v8f acc = {};
    acc = gemm_regB(As, bw1, lane, acc);
#pragma unroll
    for (int i = 0; i < 8; ++i) Ts[(rbase + i) * DD + col] = acc[i] + bb1;
    __syncthreads();
    row_stats(Ts, mu, rs, tid);
    __syncthreads();
    { // LN + ReLU -> bf16 A tile for second GEMM
      float m = mu[row], r = rs[row];
#pragma unroll
      for (int j = 0; j < 8; ++j) {
        int c = seg * 8 + j;
        float v = (Ts[row * DD + c] - m) * r * g1[c] + be1[c];
        v = v > 0.f ? v : 0.f;
        A2[row * DD + c] = (bf16_t)v;
      }
    }
    __syncthreads();

    v8f acc2 = {};
    acc2 = gemm_regB(A2, bw2, lane, acc2);
#pragma unroll
    for (int i = 0; i < 8; ++i) {
      int rr = tBase + rbase + i;
      if (rr < N) h_nodes[(size_t)rr * DD + col] = acc2[i] + bb2;
    }
  }
}

// ---------------------------------------------------------------------------
// Edge MLP + scatter-add.  ef = [x[src], x[dst], edge_attr] (261 wide);
// first 256 columns via two bf16 WMMA chains (register-resident B), the
// 5-wide edge_attr tail in fp32 VALU.  4 tiles (64 edges) per block.
// ---------------------------------------------------------------------------
__global__ __launch_bounds__(256) void edge_mlp_kernel(
    const bf16_t* __restrict__ x_bf, const float* __restrict__ edge_attr,
    const long long* __restrict__ eidx,  // [2, E] int64
    const bf16_t* __restrict__ w1apk, const bf16_t* __restrict__ w1bpk,
    const float* __restrict__ w1c,       // edge_w1 rows 256..260 (5 x 128 fp32)
    const float* __restrict__ b1, const float* __restrict__ g1,
    const float* __restrict__ be1,
    const bf16_t* __restrict__ w2pk, const float* __restrict__ b2,
    float* __restrict__ h_agg, int E) {
  __shared__ __align__(16) bf16_t As[16 * DD];
  __shared__ __align__(16) bf16_t Ad[16 * DD];
  __shared__ __align__(16) float  Ts[16 * DD];
  __shared__ __align__(16) bf16_t A2[16 * DD];
  __shared__ float Wc[5 * DD];
  __shared__ float EA[RPB * 5];
  __shared__ int   sIdx[RPB], dIdx[RPB];
  __shared__ float mu[16], rs[16];

  const int tid  = threadIdx.x;
  const int lane = tid & 31, wave = tid >> 5;
  const int row  = tid >> 4, seg  = tid & 15;
  const int col   = wave * 16 + (lane & 15);
  const int rbase = (lane >> 4) * 8;
  const long long eBase = (long long)blockIdx.x * RPB;

  // ---- once-per-block staging: indices, edge_attr, tail weights, B-frags
  if (tid < RPB) {
    long long e = eBase + tid;
    long long s = 0, d = 0;
    if (e < E) { s = eidx[e]; d = eidx[(long long)E + e]; }
    sIdx[tid] = (int)s;
    dIdx[tid] = (int)d;
  }
  for (int i = tid; i < RPB * 5; i += 256) {
    long long e = eBase + i / 5;
    EA[i] = (e < E) ? edge_attr[e * 5 + (i % 5)] : 0.f;
  }
  if (tid < 128) {
#pragma unroll
    for (int q = 0; q < 5; ++q) Wc[q * DD + tid] = w1c[q * DD + tid];
  }

  v16bf bw1a[KC], bw1b[KC], bw2[KC];
  load_bregs(w1apk, lane, wave, bw1a);
  load_bregs(w1bpk, lane, wave, bw1b);
  load_bregs(w2pk,  lane, wave, bw2);

  const float bb1 = b1[col];
  const float bb2 = b2[col];

  for (int t = 0; t < MT; ++t) {
    const int rBase = t * 16;
    __syncthreads();   // staging complete / previous tile LDS dead
    { // gather endpoint features (bf16, L2-resident table)
      *(uint4*)(As + row * DD + seg * 8) =
          *(const uint4*)(x_bf + (size_t)sIdx[rBase + row] * DD + seg * 8);
      *(uint4*)(Ad + row * DD + seg * 8) =
          *(const uint4*)(x_bf + (size_t)dIdx[rBase + row] * DD + seg * 8);
    }
    __syncthreads();

    v8f acc = {};
    acc = gemm_regB(As, bw1a, lane, acc);
    acc = gemm_regB(Ad, bw1b, lane, acc);
    {
      float wc0 = Wc[col], wc1 = Wc[DD + col], wc2 = Wc[2 * DD + col];
      float wc3 = Wc[3 * DD + col], wc4 = Wc[4 * DD + col];
#pragma unroll
      for (int i = 0; i < 8; ++i) {
        int rr = rBase + rbase + i;
        float extra = EA[rr * 5 + 0] * wc0 + EA[rr * 5 + 1] * wc1 +
                      EA[rr * 5 + 2] * wc2 + EA[rr * 5 + 3] * wc3 +
                      EA[rr * 5 + 4] * wc4;
        Ts[(rbase + i) * DD + col] = acc[i] + bb1 + extra;
      }
    }
    __syncthreads();
    row_stats(Ts, mu, rs, tid);
    __syncthreads();
    {
      float m = mu[row], r = rs[row];
#pragma unroll
      for (int j = 0; j < 8; ++j) {
        int c = seg * 8 + j;
        float v = (Ts[row * DD + c] - m) * r * g1[c] + be1[c];
        v = v > 0.f ? v : 0.f;
        A2[row * DD + c] = (bf16_t)v;
      }
    }
    __syncthreads();

    v8f acc2 = {};
    acc2 = gemm_regB(A2, bw2, lane, acc2);
#pragma unroll
    for (int i = 0; i < 8; ++i) {
      long long e = eBase + rBase + rbase + i;
      if (e < E) {
        atomicAdd(&h_agg[(size_t)dIdx[rBase + rbase + i] * DD + col],
                  acc2[i] + bb2);
      }
    }
  }
}

// ---------------------------------------------------------------------------
// Update: LN( [h_nodes, h_agg] @ upd_w + upd_b + x )
// ---------------------------------------------------------------------------
__global__ __launch_bounds__(256) void update_kernel(
    const float* __restrict__ h_nodes, const float* __restrict__ h_agg,
    const bf16_t* __restrict__ wuapk, const bf16_t* __restrict__ wubpk,
    const float* __restrict__ ub, const float* __restrict__ x,
    const float* __restrict__ lng, const float* __restrict__ lnb,
    float* __restrict__ out, int N) {
  __shared__ __align__(16) bf16_t An[16 * DD];
  __shared__ __align__(16) bf16_t Aa[16 * DD];
  __shared__ __align__(16) float  Ts[16 * DD];
  __shared__ float mu[16], rs[16];

  const int tid  = threadIdx.x;
  const int lane = tid & 31, wave = tid >> 5;
  const int row  = tid >> 4, seg  = tid & 15;
  const int col   = wave * 16 + (lane & 15);
  const int rbase = (lane >> 4) * 8;
  const int nBase = blockIdx.x * RPB;

  v16bf bwa[KC], bwb[KC];
  load_bregs(wuapk, lane, wave, bwa);
  load_bregs(wubpk, lane, wave, bwb);
  const float bbu = ub[col];

  for (int t = 0; t < MT; ++t) {
    const int tBase = nBase + t * 16;
    const int gr = tBase + row;
    __syncthreads();
    {
      uint4 un, ua;
      bf16_t* hn = (bf16_t*)&un;
      bf16_t* ha = (bf16_t*)&ua;
      if (gr < N) {
        const float* pn = h_nodes + (size_t)gr * DD + seg * 8;
        const float* pa = h_agg   + (size_t)gr * DD + seg * 8;
#pragma unroll
        for (int j = 0; j < 8; ++j) { hn[j] = (bf16_t)pn[j]; ha[j] = (bf16_t)pa[j]; }
      } else {
#pragma unroll
        for (int j = 0; j < 8; ++j) { hn[j] = (bf16_t)0.0f; ha[j] = (bf16_t)0.0f; }
      }
      *(uint4*)(An + row * DD + seg * 8) = un;
      *(uint4*)(Aa + row * DD + seg * 8) = ua;
    }
    __syncthreads();

    v8f acc = {};
    acc = gemm_regB(An, bwa, lane, acc);
    acc = gemm_regB(Aa, bwb, lane, acc);
#pragma unroll
    for (int i = 0; i < 8; ++i) {
      int rr = tBase + rbase + i;
      float resid = (rr < N) ? x[(size_t)rr * DD + col] : 0.f;
      Ts[(rbase + i) * DD + col] = acc[i] + bbu + resid;
    }
    __syncthreads();
    row_stats(Ts, mu, rs, tid);
    __syncthreads();
    if (gr < N) {
      float m = mu[row], r = rs[row];
#pragma unroll
      for (int j = 0; j < 8; ++j) {
        int c = seg * 8 + j;
        out[(size_t)gr * DD + c] = (Ts[row * DD + c] - m) * r * lng[c] + lnb[c];
      }
    }
  }
}

// ---------------------------------------------------------------------------
extern "C" void kernel_launch(void* const* d_in, const int* in_sizes, int n_in,
                              void* d_out, int out_size, void* d_ws, size_t ws_size,
                              hipStream_t stream) {
  const float* x        = (const float*)d_in[0];
  const float* edge_attr= (const float*)d_in[1];
  // d_in[2] = pos (unused by reference)
  const float* node_w1  = (const float*)d_in[3];
  const float* node_b1  = (const float*)d_in[4];
  const float* node_g1  = (const float*)d_in[5];
  const float* node_be1 = (const float*)d_in[6];
  const float* node_w2  = (const float*)d_in[7];
  const float* node_b2  = (const float*)d_in[8];
  const float* edge_w1  = (const float*)d_in[9];
  const float* edge_b1  = (const float*)d_in[10];
  const float* edge_g1  = (const float*)d_in[11];
  const float* edge_be1 = (const float*)d_in[12];
  const float* edge_w2  = (const float*)d_in[13];
  const float* edge_b2  = (const float*)d_in[14];
  const float* upd_w    = (const float*)d_in[15];
  const float* upd_b    = (const float*)d_in[16];
  const float* ln_g     = (const float*)d_in[17];
  const float* ln_b     = (const float*)d_in[18];
  const long long* eidx = (const long long*)d_in[19];

  const int N = in_sizes[0] / DD;   // x is (N, 128)
  const int E = in_sizes[1] / 5;    // edge_attr is (E, 5)

  // workspace carve-up (256B aligned)
  char* w = (char*)d_ws;
  auto take = [&](size_t bytes) -> void* {
    void* p = (void*)w;
    w += (bytes + 255) & ~(size_t)255;
    return p;
  };
  bf16_t* x_bf    = (bf16_t*)take((size_t)N * DD * sizeof(bf16_t));
  float*  h_nodes = (float*) take((size_t)N * DD * sizeof(float));
  float*  h_agg   = (float*) take((size_t)N * DD * sizeof(float));
  bf16_t* packed  = (bf16_t*)take((size_t)7 * 16384 * sizeof(bf16_t));

  bf16_t* pk_nw1  = packed + 0 * 16384;
  bf16_t* pk_nw2  = packed + 1 * 16384;
  bf16_t* pk_ew1a = packed + 2 * 16384;
  bf16_t* pk_ew1b = packed + 3 * 16384;
  bf16_t* pk_ew2  = packed + 4 * 16384;
  bf16_t* pk_ua   = packed + 5 * 16384;
  bf16_t* pk_ub   = packed + 6 * 16384;

  hipMemsetAsync(h_agg, 0, (size_t)N * DD * sizeof(float), stream);

  pack_weights_kernel<<<(7 * 16384) / 256, 256, 0, stream>>>(
      node_w1, node_w2, edge_w1, edge_w2, upd_w, packed);

  node_mlp_kernel<<<(N + RPB - 1) / RPB, 256, 0, stream>>>(
      x, pk_nw1, node_b1, node_g1, node_be1, pk_nw2, node_b2, h_nodes, x_bf, N);

  edge_mlp_kernel<<<(E + RPB - 1) / RPB, 256, 0, stream>>>(
      x_bf, edge_attr, eidx, pk_ew1a, pk_ew1b, edge_w1 + 256 * DD,
      edge_b1, edge_g1, edge_be1, pk_ew2, edge_b2, h_agg, E);

  update_kernel<<<(N + RPB - 1) / RPB, 256, 0, stream>>>(
      h_nodes, h_agg, pk_ua, pk_ub, upd_b, x, ln_g, ln_b, (float*)d_out, N);
}
